// Attention_89867895702141
// MI455X (gfx1250) — matrix operand order repaired and checked by reading, not compiled, not run on hardware
//
#include <hip/hip_runtime.h>

#define NHEADS   32
#define NKVH     8
#define HD       128
#define BATCH    4
#define SEQ      1024
#define STOT     (BATCH * SEQ)
#define GQA      4
#define NUM_SLOTS 8192
#define KVROW    (NKVH * HD)   // 1024
#define QROW     (NHEADS * HD) // 4096

// SCALE * log2(e), folded so softmax uses exp2 (v_exp_f32)
#define SCLOG2 (0.08838834764831845f * 1.4426950408889634f)
#define NEGINF (-3.0e38f)

typedef __attribute__((ext_vector_type(16))) __bf16 v16bf;
typedef __attribute__((ext_vector_type(8)))  __bf16 v8bf;
typedef __attribute__((ext_vector_type(8)))  float  v8f;

__device__ inline v8bf cvt8(const float* __restrict__ p) {
  const float4 a = *(const float4*)p;
  const float4 b = *(const float4*)(p + 4);
  v8bf r;
  r[0] = (__bf16)a.x; r[1] = (__bf16)a.y; r[2] = (__bf16)a.z; r[3] = (__bf16)a.w;
  r[4] = (__bf16)b.x; r[5] = (__bf16)b.y; r[6] = (__bf16)b.z; r[7] = (__bf16)b.w;
  return r;
}

__device__ inline v16bf cat16(v8bf lo, v8bf hi) {
  v16bf r;
#pragma unroll
  for (int i = 0; i < 8; ++i) { r[i] = lo[i]; r[i + 8] = hi[i]; }
  return r;
}

// 16 consecutive bf16 from LDS (32B-aligned) -> two ds_load_b128
__device__ inline v16bf lds16(const __bf16* p) {
  return cat16(*(const v8bf*)p, *(const v8bf*)(p + 8));
}

__device__ inline v8f wmma_bf16(v16bf a, v16bf b, v8f c) {
  // (neg_a, A, neg_b, B, c_mod, C, reuse_a, reuse_b)
  return __builtin_amdgcn_wmma_f32_16x16x32_bf16(false, a, false, b, (short)0, c,
                                                 false, false);
}

// ---------------------------------------------------------------------------
// Intra-row (16-lane) XOR butterfly via v_permlane16_b32 (VALU), keeping the
// softmax reductions off the LDS pipe. Selector: lane i reads nibble i of
// {S2,S1}; rows of 16 lanes are permuted independently.
// ---------------------------------------------------------------------------
__device__ inline float permxor16(float x, int s1, int s2) {
  return __int_as_float(__builtin_amdgcn_permlane16(
      __float_as_int(x), __float_as_int(x), s1, s2, false, false));
}

__device__ inline float rowmax16(float x) {
  x = fmaxf(x, permxor16(x, 0x67452301, 0xEFCDAB89));   // xor 1
  x = fmaxf(x, permxor16(x, 0x54761032, 0xDCFE98BA));   // xor 2
  x = fmaxf(x, permxor16(x, 0x32107654, 0xBA98FEDC));   // xor 4
  x = fmaxf(x, permxor16(x, 0xFEDCBA98, 0x76543210));   // xor 8
  return x;
}

__device__ inline float rowsum16(float x) {
  x += permxor16(x, 0x67452301, 0xEFCDAB89);            // xor 1
  x += permxor16(x, 0x54761032, 0xDCFE98BA);            // xor 2
  x += permxor16(x, 0x32107654, 0xBA98FEDC);            // xor 4
  x += permxor16(x, 0xFEDCBA98, 0x76543210);            // xor 8
  return x;
}

// ---------------------------------------------------------------------------
// Flash-attention prefill (causal, GQA). One block = (b, head, 128-row qblock),
// 8 waves, each wave owns 16 query rows. K/V tiles of 32 tokens staged in LDS.
// ---------------------------------------------------------------------------
__global__ __launch_bounds__(256, 1)
void fa_prefill_kernel(const float* __restrict__ q,
                       const float* __restrict__ k,
                       const float* __restrict__ v,
                       float* __restrict__ out) {
  __shared__ __bf16 Klds[32][HD];      // [token][feature]   8 KB
  __shared__ __bf16 Vt[HD][32];        // [feature][token]   8 KB (transposed)
  __shared__ __bf16 Plds[8][16][32];   // per-wave P scratch 8 KB

  const int tid  = threadIdx.x;
  const int lane = tid & 31;
  const int w    = tid >> 5;           // wave id 0..7
  const int ln16 = lane & 15;
  const int grp  = lane >> 4;          // 16-lane group (A/B layout half)

  const int bid  = blockIdx.x;
  const int qblk = bid & 7;            // 8 q-blocks of 128 rows
  const int h    = (bid >> 3) & 31;    // query head
  const int b    = bid >> 8;           // batch
  const int kvh  = h >> 2;             // GQA: 4 q heads per kv head

  const int qrow0 = qblk * 128 + w * 16;   // wave's first query row (in-seq)

  // ---- load Q tile (16 x 128) as 4 bf16 A-fragments (16x32 each) ----------
  // A fragment as packed here (16x32, contiguous k): lanes row M=ln16 hold K run {16*grp..+15}
  v16bf aq[4];
  {
    const float* qp = q + (size_t)(b * SEQ + qrow0 + ln16) * QROW + h * HD;
#pragma unroll
    for (int c = 0; c < 4; ++c) {
      const int f1 = c * 32 + 16 * grp;
      const int f2 = c * 32 + 16 * grp + 8;
      aq[c] = cat16(cvt8(qp + f1), cvt8(qp + f2));
    }
  }

  v8f oacc[8];
#pragma unroll
  for (int nb = 0; nb < 8; ++nb) oacc[nb] = (v8f)0.0f;
  float mr[8], lr[8];
#pragma unroll
  for (int r = 0; r < 8; ++r) { mr[r] = NEGINF; lr[r] = 0.0f; }

  const int nTilesBlock = qblk * 4 + 4;             // block causal bound
  // Wave-uniform causal bound: force into an SGPR so the guard below becomes
  // a scalar branch and EXEC stays all-ones around the WMMAs (ISA 7.12 req).
  const int nTilesWave =
      __builtin_amdgcn_readfirstlane((qrow0 + 15) / 32 + 1);

  for (int t = 0; t < nTilesBlock; ++t) {
    const int kbase = t * 32;                       // tile token base (in-seq)
    __syncthreads();                                // LDS reuse fence

    // ---- cooperative staging: 256 threads, 16 features x 1 token each ----
    {
      const int tt = tid >> 3;                      // token 0..31
      const int f0 = (tid & 7) * 16;                // feature base
      const size_t row = (size_t)(b * SEQ + kbase + tt) * KVROW + kvh * HD + f0;
      const float* kp = k + row;
      const float* vp = v + row;

      // Speculative prefetch of the next tile (global_prefetch_b8): pull the
      // next 32 tokens toward GL2 while this tile is being computed.
      if (t + 1 < nTilesBlock) {
        __builtin_prefetch(kp + (size_t)32 * KVROW, 0, 1);
        __builtin_prefetch(vp + (size_t)32 * KVROW, 0, 1);
      }

      *(v8bf*)&Klds[tt][f0]     = cvt8(kp);
      *(v8bf*)&Klds[tt][f0 + 8] = cvt8(kp + 8);
#pragma unroll
      for (int j = 0; j < 16; ++j)                  // transposed V store
        Vt[f0 + j][tt] = (__bf16)vp[j];
    }
    __syncthreads();

    if (t < nTilesWave) {                           // scalar (SGPR) guard
      // ---- S = Q * K^T : two 16x16 subtiles over 4 feature chunks --------
      v8f s0 = (v8f)0.0f, s1 = (v8f)0.0f;
#pragma unroll
      for (int c = 0; c < 4; ++c) {
        // B as loaded here (32x16, same contiguous k as A): lane = N + 16*(K/16); N = token, K = feature
        v16bf kb0 = lds16(&Klds[ln16][c * 32 + 16 * grp]);
        v16bf kb1 = lds16(&Klds[16 + ln16][c * 32 + 16 * grp]);
        s0 = wmma_bf16(aq[c], kb0, s0);
        s1 = wmma_bf16(aq[c], kb1, s1);
      }

      // ---- online softmax on C-layout f32 regs ---------------------------
#pragma unroll
      for (int r = 0; r < 8; ++r) {
        const int qg = qrow0 + r + 8 * grp;         // this row's query index
        float e0 = s0[r] * SCLOG2;
        float e1 = s1[r] * SCLOG2;
        if (kbase + ln16 > qg)      e0 = NEGINF;    // causal mask
        if (kbase + 16 + ln16 > qg) e1 = NEGINF;

        const float tmax  = rowmax16(fmaxf(e0, e1));
        const float mnew  = fmaxf(mr[r], tmax);
        const float alpha = exp2f(mr[r] - mnew);
        const float p0 = exp2f(e0 - mnew);
        const float p1 = exp2f(e1 - mnew);

        const float rs = rowsum16(p0 + p1);

        lr[r] = lr[r] * alpha + rs;
        mr[r] = mnew;
#pragma unroll
        for (int nb = 0; nb < 8; ++nb) oacc[nb][r] *= alpha;

        // C layout -> LDS (row = r + 8*grp, col = token within tile)
        const int row = r + 8 * grp;
        Plds[w][row][ln16]      = (__bf16)p0;
        Plds[w][row][16 + ln16] = (__bf16)p1;
      }

      // same-wave LDS ordering before re-reading P in A layout
      asm volatile("s_wait_dscnt 0x0" ::: "memory");

      // P as A-fragment 16x32 (K dim = 32 tokens)
      v8bf plo = *(const v8bf*)&Plds[w][ln16][16 * grp];
      v8bf phi = *(const v8bf*)&Plds[w][ln16][16 * grp + 8];
      v16bf ap = cat16(plo, phi);

      // ---- O += P * V : 8 feature blocks of 16 --------------------------
#pragma unroll
      for (int nb = 0; nb < 8; ++nb) {
        // B as loaded here (same contiguous k as P): lane = feature(N) + 16*(token/16); Vt rows are features
        v16bf vb = lds16(&Vt[nb * 16 + ln16][16 * grp]);
        oacc[nb] = wmma_bf16(ap, vb, oacc[nb]);
      }
    }
  }

  // ---- epilogue: normalize and store ---------------------------------------
#pragma unroll
  for (int r = 0; r < 8; ++r) {
    const float inv = 1.0f / lr[r];
    const size_t orow =
        (size_t)(b * SEQ + qrow0 + r + 8 * grp) * QROW + h * HD;
#pragma unroll
    for (int nb = 0; nb < 8; ++nb)
      out[orow + nb * 16 + ln16] = oacc[nb][r] * inv;
  }
}

// ---------------------------------------------------------------------------
// KV-cache: copy input caches into d_out (it is poisoned), then scatter rows.
// ---------------------------------------------------------------------------
__global__ void cache_copy_kernel(const float* __restrict__ kc_in,
                                  const float* __restrict__ vc_in,
                                  float* __restrict__ kc_out,
                                  float* __restrict__ vc_out, int n) {
  for (int i = blockIdx.x * blockDim.x + threadIdx.x; i < n;
       i += gridDim.x * blockDim.x) {
    kc_out[i] = kc_in[i];
    vc_out[i] = vc_in[i];
  }
}

__global__ void cache_scatter_kernel(const float* __restrict__ k,
                                     const float* __restrict__ v,
                                     const int* __restrict__ slot_mapping,
                                     float* __restrict__ kc_out,
                                     float* __restrict__ vc_out, int n) {
  for (int i = blockIdx.x * blockDim.x + threadIdx.x; i < n;
       i += gridDim.x * blockDim.x) {
    const int s   = i >> 10;          // KVROW == 1024
    const int rem = i & 1023;
    const size_t dst = (size_t)slot_mapping[s] * KVROW + rem;
    kc_out[dst] = k[i];
    vc_out[dst] = v[i];
  }
}

extern "C" void kernel_launch(void* const* d_in, const int* in_sizes, int n_in,
                              void* d_out, int out_size, void* d_ws,
                              size_t ws_size, hipStream_t stream) {
  (void)in_sizes; (void)n_in; (void)out_size; (void)d_ws; (void)ws_size;

  const float* q       = (const float*)d_in[0];
  const float* k       = (const float*)d_in[1];
  const float* v       = (const float*)d_in[2];
  const float* kc_in   = (const float*)d_in[3];
  const float* vc_in   = (const float*)d_in[4];
  const int*   slotmap = (const int*)d_in[5];

  float* out    = (float*)d_out;
  float* kc_out = out + (size_t)STOT * QROW;                 // after O
  float* vc_out = kc_out + (size_t)NUM_SLOTS * KVROW;        // after k_cache

  // 1) bring caches to known state (reference caches are zero-filled inputs)
  {
    const int n = NUM_SLOTS * KVROW;                         // 8,388,608
    cache_copy_kernel<<<2048, 256, 0, stream>>>(kc_in, vc_in, kc_out, vc_out, n);
  }
  // 2) scatter current K/V rows into cache slots
  {
    const int n = STOT * KVROW;                              // 4,194,304
    cache_scatter_kernel<<<2048, 256, 0, stream>>>(k, v, slotmap, kc_out,
                                                   vc_out, n);
  }
  // 3) causal GQA flash attention
  {
    dim3 grid(BATCH * NHEADS * (SEQ / 128));                 // 1024 blocks
    fa_prefill_kernel<<<grid, 256, 0, stream>>>(q, k, v, out);
  }
}
